// GraphPro_68289980006751
// MI455X (gfx1250) — matrix operand order repaired and checked by reading, not compiled, not run on hardware
//
#include <hip/hip_runtime.h>
#include <hip/hip_bf16.h>

typedef __attribute__((ext_vector_type(16))) __bf16 v16bf;
typedef __attribute__((ext_vector_type(8)))  __bf16 v8bf;
typedef __attribute__((ext_vector_type(8)))  float  v8f;

#define Bn 4
#define Nn 320
#define Cc 128
#define H2 192
#define H1 96
#define EPSK 1e-5f
#define SLOPE 0.01f
#define NT 256
#define NW 8

// ---- LDS byte offsets (all 32B aligned) ----
#define W1OFF   0        // 192*128 bf16 = 49152 B
#define W2OFF   49152    // 192*192 bf16 = 73728 B
#define W3OFF   122880   //  96*192 bf16 = 36864 B
#define W4OFF   159744   //  96* 96 bf16 = 18432 B
#define SCROFF  178176   // 8 waves * 16*192 bf16 = 49152 B
#define LGOFF   227328   // 320 f32
#define XAOFF   228608   // 128 f32
#define W5OFF   229120   // 97 f32 (W5 + b5)
#define BNSCOFF 229536   // 576 f32 scales
#define BNSHOFF 231840   // 576 f32 shifts
#define LDS_TOTAL 234176

static __device__ __forceinline__ v8f wmma_bf16(v16bf a, v16bf b, v8f c) {
    return __builtin_amdgcn_wmma_f32_16x16x32_bf16(false, a, false, b, (short)0, c, false, false);
}

// |x| via v_max_num_f32(x, -x): opaque to SLP/DAG combines (works around a
// gfx1250 backend crash selecting vectorized fabs-of-fsub).
static __device__ __forceinline__ float absf_hw(float x) {
    float r;
    asm("v_max_num_f32 %0, %1, -%1" : "=v"(r) : "v"(x));
    return r;
}

// A-fragment (16x32 bf16, M=lane&15, K-half by lane>>4): two 16B groups 16 ch apart
static __device__ __forceinline__ v16bf load_afrag(const __bf16* p) {
    v8bf lo = *(const v8bf*)p;
    v8bf hi = *(const v8bf*)(p + 16);
    return __builtin_shufflevector(lo, hi, 0,1,2,3,4,5,6,7,8,9,10,11,12,13,14,15);
}

__global__ __launch_bounds__(NT) void graphpro_fused(
    const float* __restrict__ X,
    const float* __restrict__ W1g, const float* __restrict__ W2g,
    const float* __restrict__ W3g, const float* __restrict__ W4g,
    const float* __restrict__ W5g, const float* __restrict__ b5g,
    const float* __restrict__ Wp,  const float* __restrict__ bpv,
    const float* __restrict__ g1, const float* __restrict__ bb1, const float* __restrict__ m1, const float* __restrict__ v1,
    const float* __restrict__ g2, const float* __restrict__ bb2, const float* __restrict__ m2, const float* __restrict__ v2,
    const float* __restrict__ g3, const float* __restrict__ bb3, const float* __restrict__ m3, const float* __restrict__ v3,
    const float* __restrict__ g4, const float* __restrict__ bb4, const float* __restrict__ m4, const float* __restrict__ v4,
    float* __restrict__ out)
{
    extern __shared__ __align__(32) char smem[];
    __bf16* w1   = (__bf16*)(smem + W1OFF);
    __bf16* w2   = (__bf16*)(smem + W2OFF);
    __bf16* w3   = (__bf16*)(smem + W3OFF);
    __bf16* w4   = (__bf16*)(smem + W4OFF);
    float*  lgs  = (float*)(smem + LGOFF);
    float*  xa   = (float*)(smem + XAOFF);
    float*  w5s  = (float*)(smem + W5OFF);
    float*  bnsc = (float*)(smem + BNSCOFF);
    float*  bnsh = (float*)(smem + BNSHOFF);

    const int tid = threadIdx.x;
    const int wg  = blockIdx.x;
    const int b   = wg / Nn;
    const int i   = wg % Nn;

    // ---- stage weights (f32 -> bf16) into LDS ----
    for (int t = tid; t < H2*Cc; t += NT) w1[t] = (__bf16)W1g[t];
    for (int t = tid; t < H2*H2; t += NT) w2[t] = (__bf16)W2g[t];
    for (int t = tid; t < H1*H2; t += NT) w3[t] = (__bf16)W3g[t];
    for (int t = tid; t < H1*H1; t += NT) w4[t] = (__bf16)W4g[t];
    if (tid < H1)  w5s[tid] = W5g[tid];
    if (tid == H1) w5s[H1]  = b5g[0];
    // ---- fold BN into per-channel scale/shift ----
    for (int t = tid; t < 576; t += NT) {
        const float *g, *bb, *mm, *vv; int c;
        if (t < 192)      { g = g1; bb = bb1; mm = m1; vv = v1; c = t; }
        else if (t < 384) { g = g2; bb = bb2; mm = m2; vv = v2; c = t - 192; }
        else if (t < 480) { g = g3; bb = bb3; mm = m3; vv = v3; c = t - 384; }
        else              { g = g4; bb = bb4; mm = m4; vv = v4; c = t - 480; }
        float sc = g[c] * rsqrtf(vv[c] + EPSK);
        bnsc[t] = sc;
        bnsh[t] = bb[c] - mm[c] * sc;
    }
    __syncthreads();

    const int lane  = tid & 31;
    const int wave  = tid >> 5;
    const int lrow  = lane & 15;   // A-row / B-col / D-col within 16-tile
    const int khalf = lane >> 4;   // K-half selector
    __bf16* scr = (__bf16*)(smem + SCROFF) + wave * (16 * H2);
    const float* xi = X + ((size_t)(b * Nn + i)) * Cc;

    for (int jt = wave; jt < Nn / 16; jt += NW) {
        const float* xj = X + ((size_t)(b * Nn + jt * 16 + lrow)) * Cc;

        // ---- layer-1 A fragments straight from global: a = |xj - xi| in bf16 ----
        v16bf a1[4];
        #pragma unroll
        for (int kc = 0; kc < 4; ++kc) {
            const int c0 = kc * 32 + khalf * 8;
            const int c1 = c0 + 16;
            v16bf a;
            #pragma unroll
            for (int e = 0; e < 8; ++e) {
                a[e]     = (__bf16)absf_hw(xj[c0 + e] - xi[c0 + e]);
                a[e + 8] = (__bf16)absf_hw(xj[c1 + e] - xi[c1 + e]);
            }
            a1[kc] = a;
        }

        // ---- layer 1: 128 -> 192 ----
        #pragma unroll
        for (int nt = 0; nt < H2 / 16; ++nt) {
            v8f d = {};
            #pragma unroll
            for (int kc = 0; kc < 4; ++kc) {
                v16bf bf = load_afrag(w1 + (nt * 16 + lrow) * Cc + kc * 32 + khalf * 8);
                d = wmma_bf16(a1[kc], bf, d);
            }
            const int c = nt * 16 + lrow;
            const float sc = bnsc[c], sh = bnsh[c];
            #pragma unroll
            for (int v = 0; v < 8; ++v) {
                float x = d[v] * sc + sh;
                x = x >= 0.f ? x : SLOPE * x;
                scr[(khalf * 8 + v) * H2 + c] = (__bf16)x;
            }
        }
        asm volatile("s_wait_dscnt 0" ::: "memory");

        // ---- layer 2: 192 -> 192 ----
        v16bf a2[6];
        #pragma unroll
        for (int kc = 0; kc < 6; ++kc)
            a2[kc] = load_afrag(scr + lrow * H2 + kc * 32 + khalf * 8);
        #pragma unroll
        for (int nt = 0; nt < H2 / 16; ++nt) {
            v8f d = {};
            #pragma unroll
            for (int kc = 0; kc < 6; ++kc) {
                v16bf bf = load_afrag(w2 + (nt * 16 + lrow) * H2 + kc * 32 + khalf * 8);
                d = wmma_bf16(a2[kc], bf, d);
            }
            const int c = nt * 16 + lrow;
            const float sc = bnsc[192 + c], sh = bnsh[192 + c];
            #pragma unroll
            for (int v = 0; v < 8; ++v) {
                float x = d[v] * sc + sh;
                x = x >= 0.f ? x : SLOPE * x;
                scr[(khalf * 8 + v) * H2 + c] = (__bf16)x;
            }
        }
        asm volatile("s_wait_dscnt 0" ::: "memory");

        // ---- layer 3: 192 -> 96 ----
        v16bf a3[6];
        #pragma unroll
        for (int kc = 0; kc < 6; ++kc)
            a3[kc] = load_afrag(scr + lrow * H2 + kc * 32 + khalf * 8);
        #pragma unroll
        for (int nt = 0; nt < H1 / 16; ++nt) {
            v8f d = {};
            #pragma unroll
            for (int kc = 0; kc < 6; ++kc) {
                v16bf bf = load_afrag(w3 + (nt * 16 + lrow) * H2 + kc * 32 + khalf * 8);
                d = wmma_bf16(a3[kc], bf, d);
            }
            const int c = nt * 16 + lrow;
            const float sc = bnsc[384 + c], sh = bnsh[384 + c];
            #pragma unroll
            for (int v = 0; v < 8; ++v) {
                float x = d[v] * sc + sh;
                x = x >= 0.f ? x : SLOPE * x;
                scr[(khalf * 8 + v) * H2 + c] = (__bf16)x;
            }
        }
        asm volatile("s_wait_dscnt 0" ::: "memory");

        // ---- layer 4: 96 -> 96 ----
        v16bf a4[3];
        #pragma unroll
        for (int kc = 0; kc < 3; ++kc)
            a4[kc] = load_afrag(scr + lrow * H2 + kc * 32 + khalf * 8);
        #pragma unroll
        for (int nt = 0; nt < H1 / 16; ++nt) {
            v8f d = {};
            #pragma unroll
            for (int kc = 0; kc < 3; ++kc) {
                v16bf bf = load_afrag(w4 + (nt * 16 + lrow) * H1 + kc * 32 + khalf * 8);
                d = wmma_bf16(a4[kc], bf, d);
            }
            const int c = nt * 16 + lrow;
            const float sc = bnsc[480 + c], sh = bnsh[480 + c];
            #pragma unroll
            for (int v = 0; v < 8; ++v) {
                float x = d[v] * sc + sh;
                x = x >= 0.f ? x : SLOPE * x;
                scr[(khalf * 8 + v) * H2 + c] = (__bf16)x;
            }
        }
        asm volatile("s_wait_dscnt 0" ::: "memory");

        // ---- layer 5: 96 -> 1 (logit per row) ----
        if (lane < 16) {
            float s = w5s[H1]; // b5
            #pragma unroll 8
            for (int c = 0; c < H1; ++c)
                s += (float)scr[lane * H2 + c] * w5s[c];
            lgs[jt * 16 + lane] = s;
        }
    }
    __syncthreads();

    // ---- softmax over j (320 logits) ----
    float mx = -3.402823466e38f;
    for (int j = 0; j < Nn; ++j) mx = fmaxf(mx, lgs[j]);
    float ssum = 0.f;
    for (int j = 0; j < Nn; ++j) ssum += __expf(lgs[j] - mx);
    const float inv = 1.f / ssum;
    __syncthreads();
    for (int j = tid; j < Nn; j += NT) lgs[j] = __expf(lgs[j] - mx) * inv;
    __syncthreads();

    // ---- Xa[c] = sum_j p[j] * X[b,j,c]  (f32) ----
    if (tid < Cc) {
        float acc = 0.f;
        const float* xb = X + (size_t)b * Nn * Cc + tid;
        for (int j = 0; j < Nn; ++j) acc += lgs[j] * xb[(size_t)j * Cc];
        xa[tid] = acc;
    }
    __syncthreads();

    // ---- out[b,i,c] = Xa . Wp[c,:] + bp[c] ----
    if (tid < Cc) {
        float o = bpv[tid];
        const float* wr = Wp + tid * Cc;
        #pragma unroll 8
        for (int k = 0; k < Cc; ++k) o += xa[k] * wr[k];
        out[((size_t)(b * Nn + i)) * Cc + tid] = o;
    }
}

extern "C" void kernel_launch(void* const* d_in, const int* in_sizes, int n_in,
                              void* d_out, int out_size, void* d_ws, size_t ws_size,
                              hipStream_t stream) {
    (void)in_sizes; (void)n_in; (void)d_ws; (void)ws_size; (void)out_size;
    // d_in order: A_init(unused), X, W1..W5, b5, Wp, bp, g1,bb1,m1,v1, g2.., g3.., g4..
    const float* X  = (const float*)d_in[1];
    const float* W1 = (const float*)d_in[2];
    const float* W2 = (const float*)d_in[3];
    const float* W3 = (const float*)d_in[4];
    const float* W4 = (const float*)d_in[5];
    const float* W5 = (const float*)d_in[6];
    const float* b5 = (const float*)d_in[7];
    const float* Wp = (const float*)d_in[8];
    const float* bp = (const float*)d_in[9];
    const float* g1 = (const float*)d_in[10]; const float* bb1 = (const float*)d_in[11];
    const float* m1 = (const float*)d_in[12]; const float* v1  = (const float*)d_in[13];
    const float* g2 = (const float*)d_in[14]; const float* bb2 = (const float*)d_in[15];
    const float* m2 = (const float*)d_in[16]; const float* v2  = (const float*)d_in[17];
    const float* g3 = (const float*)d_in[18]; const float* bb3 = (const float*)d_in[19];
    const float* m3 = (const float*)d_in[20]; const float* v3  = (const float*)d_in[21];
    const float* g4 = (const float*)d_in[22]; const float* bb4 = (const float*)d_in[23];
    const float* m4 = (const float*)d_in[24]; const float* v4  = (const float*)d_in[25];

    graphpro_fused<<<dim3(Bn * Nn), dim3(NT), LDS_TOTAL, stream>>>(
        X, W1, W2, W3, W4, W5, b5, Wp, bp,
        g1, bb1, m1, v1, g2, bb2, m2, v2,
        g3, bb3, m3, v3, g4, bb4, m4, v4,
        (float*)d_out);
}
